// MultiheadAttention_64553358459004
// MI455X (gfx1250) — matrix operand order repaired and checked by reading, not compiled, not run on hardware
//
#include <hip/hip_runtime.h>

typedef __attribute__((ext_vector_type(16))) _Float16 v16h;
typedef __attribute__((ext_vector_type(8)))  _Float16 v8h;
typedef __attribute__((ext_vector_type(8)))  float    v8f;

#define WMMA_F16(a, b, c) \
  __builtin_amdgcn_wmma_f32_16x16x32_f16(false, (a), false, (b), (short)0, (c), false, false)

#define ASYNC_CP(ldsOff, gptr) \
  asm volatile("global_load_async_to_lds_b128 %0, %1, off" :: "v"(ldsOff), "v"(gptr) : "memory")
#define ASYNC_CP_O16(ldsOff, gptr) \
  asm volatile("global_load_async_to_lds_b128 %0, %1, off offset:16" :: "v"(ldsOff), "v"(gptr) : "memory")
#define ASYNC_CP_O32(ldsOff, gptr) \
  asm volatile("global_load_async_to_lds_b128 %0, %1, off offset:32" :: "v"(ldsOff), "v"(gptr) : "memory")
#define ASYNC_CP_O48(ldsOff, gptr) \
  asm volatile("global_load_async_to_lds_b128 %0, %1, off offset:48" :: "v"(ldsOff), "v"(gptr) : "memory")
#define WAIT_ASYNC_0() asm volatile("s_wait_asynccnt 0x0" ::: "memory")
#define WAIT_ASYNC_4() asm volatile("s_wait_asynccnt 0x4" ::: "memory")
#define WAIT_ASYNC_8() asm volatile("s_wait_asynccnt 0x8" ::: "memory")

constexpr int B_  = 2;
constexpr int S_  = 2048;
constexpr int D_  = 1024;
constexpr int H_  = 16;
constexpr int DH_ = 64;
constexpr int M_  = B_ * S_;            // 4096
constexpr float NEG_INF_ = -1000000000.0f;

__device__ __forceinline__ uint32_t lds_off(const void* p) {
  return (uint32_t)(uintptr_t)p;        // low 32 bits of generic ptr = LDS byte offset
}

__device__ __forceinline__ v16h ld16(const _Float16* p, int o0, int o1) {
  v8h lo = *(const v8h*)(p + o0);
  v8h hi = *(const v8h*)(p + o1);
  return __builtin_shufflevector(lo, hi, 0,1,2,3,4,5,6,7,8,9,10,11,12,13,14,15);
}

// ---------------------------------------------------------------------------
// fp32 -> f16 bulk convert (n multiple of 8)
// ---------------------------------------------------------------------------
__global__ __launch_bounds__(256)
void cvt_f32_f16_kernel(const float4* __restrict__ x, v8h* __restrict__ y, int n8)
{
  int i = blockIdx.x * 256 + threadIdx.x;
  if (i < n8) {
    float4 a = x[i * 2], b = x[i * 2 + 1];
    v8h r;
    r[0] = (_Float16)a.x; r[1] = (_Float16)a.y; r[2] = (_Float16)a.z; r[3] = (_Float16)a.w;
    r[4] = (_Float16)b.x; r[5] = (_Float16)b.y; r[6] = (_Float16)b.z; r[7] = (_Float16)b.w;
    y[i] = r;
  }
}

// ---------------------------------------------------------------------------
// int32 mask -> bitmask (1 bit per element); one wave handles one word
// ---------------------------------------------------------------------------
__global__ __launch_bounds__(256)
void mask_bits_kernel(const int* __restrict__ m, uint32_t* __restrict__ bits)
{
  const int tid  = blockIdx.x * 256 + threadIdx.x;
  const int lane = threadIdx.x & 31;
  uint32_t v = (m[tid] != 0) ? (1u << lane) : 0u;
#pragma unroll
  for (int d = 16; d >= 1; d >>= 1) v |= __shfl_xor(v, d, 32);
  if (lane == 0) bits[tid >> 5] = v;
}

// ---------------------------------------------------------------------------
// f16 GEMM: C[M,1024] = A[M,1024] @ W[1024,1024]^T + bias
// block = 256 thr (8 waves), block tile 128x128, wave tile 32(M)x64(N).
// 64-deep k-steps double-buffered in LDS via global_load_async_to_lds_b128.
// mode 0: fp32 out [M,N]; 1: f16 [B,H,S,DH]; 2: f16 [B,H,DH,S]
// ---------------------------------------------------------------------------
__global__ __launch_bounds__(256)
void gemm_awt_kernel(const _Float16* __restrict__ A, const _Float16* __restrict__ W,
                     const float* __restrict__ bias, float* __restrict__ outF,
                     _Float16* __restrict__ outH, int mode)
{
  constexpr int K = 1024, N = 1024, KT = 64;
  __shared__ __align__(16) _Float16 Ab[2][128 * KT];
  __shared__ __align__(16) _Float16 Bb[2][128 * KT];

  const int tid    = threadIdx.x;
  const int lane   = tid & 31;
  const int wave   = tid >> 5;
  const int halfid = lane >> 4;
  const int ln     = lane & 15;
  const int mBase  = blockIdx.y * 128;
  const int nBase  = blockIdx.x * 128;
  const int mw     = (wave & 3) * 32;
  const int nw     = (wave >> 2) * 64;
  const int c0 = halfid * 8, c1 = 16 + halfid * 8;

  // staging: thread owns one 64B run of the A tile and one of the W tile
  const int srow = tid >> 1;            // 0..127
  const int soff = (tid & 1) * 32;      // halves
  const _Float16* gA = A + (size_t)(mBase + srow) * K + soff;
  const _Float16* gW = W + (size_t)(nBase + srow) * K + soff;
  const uint32_t lA[2] = { lds_off(&Ab[0][srow * KT + soff]), lds_off(&Ab[1][srow * KT + soff]) };
  const uint32_t lB[2] = { lds_off(&Bb[0][srow * KT + soff]), lds_off(&Bb[1][srow * KT + soff]) };

  auto issue = [&](int buf, int step) {
    const _Float16* a = gA + step * KT;
    const _Float16* w = gW + step * KT;
    ASYNC_CP    (lA[buf], a);
    ASYNC_CP_O16(lA[buf], a);
    ASYNC_CP_O32(lA[buf], a);
    ASYNC_CP_O48(lA[buf], a);
    ASYNC_CP    (lB[buf], w);
    ASYNC_CP_O16(lB[buf], w);
    ASYNC_CP_O32(lB[buf], w);
    ASYNC_CP_O48(lB[buf], w);
  };

  v8f acc[2][4];
#pragma unroll
  for (int s = 0; s < 2; ++s)
#pragma unroll
    for (int g = 0; g < 4; ++g)
#pragma unroll
      for (int i = 0; i < 8; ++i) acc[s][g][i] = 0.0f;

  issue(0, 0);
  for (int s = 0; s < K / KT; ++s) {
    if (s + 1 < K / KT) { issue((s + 1) & 1, s + 1); WAIT_ASYNC_8(); }
    else                { WAIT_ASYNC_0(); }
    __syncthreads();
    const _Float16* Abuf = Ab[s & 1];
    const _Float16* Bbuf = Bb[s & 1];

    // hoist all fragments of this k-step into distinct registers first so the
    // ds_load_b128s stay in flight and retire with descending DScnt waits
    v16h af[2][2];
#pragma unroll
    for (int sub = 0; sub < 2; ++sub)
#pragma unroll
      for (int t = 0; t < 2; ++t)
        af[sub][t] = ld16(Abuf + (mw + sub * 16 + ln) * KT + t * 32, c0, c1);
    v16h bf[4][2];
#pragma unroll
    for (int g = 0; g < 4; ++g)
#pragma unroll
      for (int t = 0; t < 2; ++t)
        bf[g][t] = ld16(Bbuf + (nw + g * 16 + ln) * KT + t * 32, c0, c1);

#pragma unroll
    for (int g = 0; g < 4; ++g)
#pragma unroll
      for (int t = 0; t < 2; ++t)
#pragma unroll
        for (int sub = 0; sub < 2; ++sub)
          acc[sub][g] = WMMA_F16(af[sub][t], bf[g][t], acc[sub][g]);
    __syncthreads();
  }

#pragma unroll
  for (int g = 0; g < 4; ++g) {
    const int n = nBase + nw + g * 16 + ln;
    const float bv = bias[n];
#pragma unroll
    for (int sub = 0; sub < 2; ++sub)
#pragma unroll
      for (int i = 0; i < 8; ++i) {
        const int m = mBase + mw + sub * 16 + i + halfid * 8;
        const float v = acc[sub][g][i] + bv;
        if (mode == 0) {
          outF[(size_t)m * N + n] = v;
        } else {
          const int b = m >> 11, s = m & (S_ - 1);
          const int h = n >> 6,  dh = n & (DH_ - 1);
          const size_t idx = (mode == 1)
              ? ((((size_t)b * H_ + h) * S_ + s) * DH_ + dh)
              : ((((size_t)b * H_ + h) * DH_ + dh) * S_ + s);
          outH[idx] = (_Float16)v;
        }
      }
  }
}

// ---------------------------------------------------------------------------
// Flash attention: block = 8 waves sharing (b, h, 128-query chunk).
// 64-key tiles double-buffered in LDS via async copies; each wave owns one
// 16-query tile, online softmax, P re-laid out via per-wave LDS.
// ---------------------------------------------------------------------------
__global__ __launch_bounds__(256)
void attn_flash_kernel(const _Float16* __restrict__ Qh,
                       const _Float16* __restrict__ Kh,
                       const _Float16* __restrict__ Vt,
                       const uint32_t* __restrict__ mbits,
                       _Float16* __restrict__ OcH)
{
  constexpr int KT = 64;                              // keys per step
  __shared__ __align__(16) _Float16 Kb[2][KT * 64];   // [krow][dh]
  __shared__ __align__(16) _Float16 Vb[2][64 * KT];   // [dh][krow]
  __shared__ __align__(16) _Float16 Pb[8][16 * KT];   // per-wave P staging

  const int tid    = threadIdx.x;
  const int lane   = tid & 31;
  const int wave   = tid >> 5;
  const int halfid = lane >> 4;
  const int ln     = lane & 15;

  const int qc = blockIdx.x & 15;             // S/128 = 16 query chunks
  const int h  = (blockIdx.x >> 4) & (H_ - 1);
  const int b  = blockIdx.x >> 8;
  const int q0 = qc * 128 + wave * 16;

  const float slope = exp2f(-0.5f * (float)(h + 1));
  const float inv_s = 0.03125f;               // 1/sqrt(1024)

  const _Float16* qbase = Qh + (((size_t)b * H_ + h) * S_ + q0) * DH_;
  const _Float16* kbase = Kh + ((size_t)b * H_ + h) * S_ * DH_;
  const _Float16* vbase = Vt + ((size_t)b * H_ + h) * DH_ * S_;
  const uint32_t* mbase = mbits + ((size_t)b * S_) * (S_ / 32);

  // staging: K tile 64x64 + V tile 64x64 (512 chunks each); 2 chunks each/thread
  const int srow = tid >> 3;                  // 0..31
  const int soff = (tid & 7) * 8;             // halves
  const _Float16* gK = kbase + (size_t)srow * DH_ + soff;
  const _Float16* gV = vbase + (size_t)srow * S_ + soff;
  const uint32_t lK[2] = { lds_off(&Kb[0][srow * 64 + soff]), lds_off(&Kb[1][srow * 64 + soff]) };
  const uint32_t lV[2] = { lds_off(&Vb[0][srow * KT + soff]), lds_off(&Vb[1][srow * KT + soff]) };

  auto issue = [&](int buf, int kt) {
    const _Float16* k0p = gK + (size_t)kt * DH_;             // K rows kt+srow, kt+srow+32
    ASYNC_CP(lK[buf],        k0p);
    ASYNC_CP(lK[buf] + 4096, k0p + 32 * DH_);
    const _Float16* v0p = gV + kt;                           // V rows srow, srow+32
    ASYNC_CP(lV[buf],        v0p);
    ASYNC_CP(lV[buf] + 4096, v0p + (size_t)32 * S_);
  };

  const int c0 = halfid * 8, c1 = 16 + halfid * 8;

  // Q A-fragments (kdim blocks dh 0..31, 32..63)
  v16h qf[2];
  {
    const _Float16* qrow = qbase + (size_t)ln * DH_;
#pragma unroll
    for (int t = 0; t < 2; ++t) qf[t] = ld16(qrow + t * 32, c0, c1);
  }

  v8f oacc[4];
#pragma unroll
  for (int g = 0; g < 4; ++g)
#pragma unroll
    for (int i = 0; i < 8; ++i) oacc[g][i] = 0.0f;

  float mrun[8], lrun[8];
#pragma unroll
  for (int i = 0; i < 8; ++i) { mrun[i] = -INFINITY; lrun[i] = 0.0f; }

  _Float16* pl = Pb[wave];

  issue(0, 0);
  for (int ki = 0; ki < S_ / KT; ++ki) {
    const int kt = ki * KT;
    if (ki + 1 < S_ / KT) { issue((ki + 1) & 1, kt + KT); WAIT_ASYNC_4(); }
    else                  { WAIT_ASYNC_0(); }
    __syncthreads();
    const _Float16* Kbuf = Kb[ki & 1];
    const _Float16* Vbuf = Vb[ki & 1];

    // ---- scores: four 16x16 tiles (64 key cols); hoist all K fragments ----
    v16h kf[4][2];
#pragma unroll
    for (int g = 0; g < 4; ++g)
#pragma unroll
      for (int t = 0; t < 2; ++t)
        kf[g][t] = ld16(Kbuf + (g * 16 + ln) * 64 + t * 32, c0, c1);

    v8f sacc[4];
#pragma unroll
    for (int g = 0; g < 4; ++g)
#pragma unroll
      for (int i = 0; i < 8; ++i) sacc[g][i] = 0.0f;
#pragma unroll
    for (int g = 0; g < 4; ++g)
#pragma unroll
      for (int t = 0; t < 2; ++t)
        sacc[g] = WMMA_F16(qf[t], kf[g][t], sacc[g]);

    // ---- ALiBi, scale, bit-mask, online softmax ----
#pragma unroll
    for (int i = 0; i < 8; ++i) {
      const int m = i + halfid * 8;
      const int q = q0 + m;
      const uint2 mw2 = *(const uint2*)(mbase + (size_t)q * (S_ / 32) + ki * 2);
      float sv[4];
#pragma unroll
      for (int g = 0; g < 4; ++g) {
        const int kc = kt + g * 16 + ln;
        float s = (sacc[g][i] - slope * fabsf((float)(q - kc))) * inv_s;
        const uint32_t w = (g < 2) ? mw2.x : mw2.y;
        const int bit = (g & 1) ? (16 + ln) : ln;
        sv[g] = ((w >> bit) & 1u) ? s : NEG_INF_;
      }
      float t = fmaxf(fmaxf(sv[0], sv[1]), fmaxf(sv[2], sv[3]));
#pragma unroll
      for (int d = 8; d >= 1; d >>= 1) t = fmaxf(t, __shfl_xor(t, d, 32));
      const float nm   = fmaxf(mrun[i], t);
      const float corr = __expf(mrun[i] - nm);
      float rs = 0.0f;
#pragma unroll
      for (int g = 0; g < 4; ++g) {
        const float e = __expf(sv[g] - nm);
        rs += e;
        pl[m * KT + g * 16 + ln] = (_Float16)e;   // P staging (same-wave LDS, in-order)
      }
#pragma unroll
      for (int d = 8; d >= 1; d >>= 1) rs += __shfl_xor(rs, d, 32);
      lrun[i] = lrun[i] * corr + rs;
      mrun[i] = nm;
#pragma unroll
      for (int g = 0; g < 4; ++g) oacc[g][i] *= corr;
    }

    // ---- P as A-fragments; hoist all V fragments ----
    v16h pf[2];
#pragma unroll
    for (int t = 0; t < 2; ++t) pf[t] = ld16(pl + ln * KT + t * 32, c0, c1);
    v16h vf[4][2];
#pragma unroll
    for (int g = 0; g < 4; ++g)
#pragma unroll
      for (int t = 0; t < 2; ++t)
        vf[g][t] = ld16(Vbuf + (g * 16 + ln) * KT + t * 32, c0, c1);

    // ---- O += P @ V ----
#pragma unroll
    for (int g = 0; g < 4; ++g)
#pragma unroll
      for (int t = 0; t < 2; ++t)
        oacc[g] = WMMA_F16(pf[t], vf[g][t], oacc[g]);
    __syncthreads();
  }

  // ---- normalize, write O as f16 [B, S, H*DH] ----
#pragma unroll
  for (int i = 0; i < 8; ++i) {
    const float inv_l = 1.0f / lrun[i];
    const int q = q0 + i + halfid * 8;
    _Float16* orow = OcH + ((size_t)b * S_ + q) * D_ + h * DH_;
#pragma unroll
    for (int g = 0; g < 4; ++g) orow[g * 16 + ln] = (_Float16)(oacc[g][i] * inv_l);
  }
}

// ---------------------------------------------------------------------------
extern "C" void kernel_launch(void* const* d_in, const int* in_sizes, int n_in,
                              void* d_out, int out_size, void* d_ws, size_t ws_size,
                              hipStream_t stream) {
  const float* embed = (const float*)d_in[0];
  const float* key   = (const float*)d_in[1];
  const int*   amask = (const int*)  d_in[2];
  const float* Wq = (const float*)d_in[3];
  const float* bq = (const float*)d_in[4];
  const float* Wk = (const float*)d_in[5];
  const float* bk = (const float*)d_in[6];
  const float* Wv = (const float*)d_in[7];
  const float* bv = (const float*)d_in[8];
  const float* Wo = (const float*)d_in[9];
  const float* bo = (const float*)d_in[10];
  float* out = (float*)d_out;

  // workspace layout (all 256B aligned by construction)
  char* ws = (char*)d_ws;
  const size_t szAct = (size_t)M_ * D_ * sizeof(_Float16);   // 8 MB
  const size_t szW   = (size_t)D_ * D_ * sizeof(_Float16);   // 2 MB
  _Float16* embedH = (_Float16*)(ws);                 ws += szAct;
  _Float16* keyH   = (_Float16*)(ws);                 ws += szAct;
  _Float16* WqH    = (_Float16*)(ws);                 ws += szW;
  _Float16* WkH    = (_Float16*)(ws);                 ws += szW;
  _Float16* WvH    = (_Float16*)(ws);                 ws += szW;
  _Float16* WoH    = (_Float16*)(ws);                 ws += szW;
  _Float16* Qh     = (_Float16*)(ws);                 ws += szAct;
  _Float16* Kh     = (_Float16*)(ws);                 ws += szAct;
  _Float16* Vt     = (_Float16*)(ws);                 ws += szAct;
  _Float16* OcH    = (_Float16*)(ws);                 ws += szAct;
  uint32_t* mbits  = (uint32_t*)(ws);                 // 1 MB

  const dim3 blk(256);

  // bulk converts to f16
  cvt_f32_f16_kernel<<<dim3(M_ * D_ / 8 / 256), blk, 0, stream>>>((const float4*)embed, (v8h*)embedH, M_ * D_ / 8);
  cvt_f32_f16_kernel<<<dim3(M_ * D_ / 8 / 256), blk, 0, stream>>>((const float4*)key,   (v8h*)keyH,   M_ * D_ / 8);
  cvt_f32_f16_kernel<<<dim3(D_ * D_ / 8 / 256), blk, 0, stream>>>((const float4*)Wq, (v8h*)WqH, D_ * D_ / 8);
  cvt_f32_f16_kernel<<<dim3(D_ * D_ / 8 / 256), blk, 0, stream>>>((const float4*)Wk, (v8h*)WkH, D_ * D_ / 8);
  cvt_f32_f16_kernel<<<dim3(D_ * D_ / 8 / 256), blk, 0, stream>>>((const float4*)Wv, (v8h*)WvH, D_ * D_ / 8);
  cvt_f32_f16_kernel<<<dim3(D_ * D_ / 8 / 256), blk, 0, stream>>>((const float4*)Wo, (v8h*)WoH, D_ * D_ / 8);

  // mask -> bitmask
  mask_bits_kernel<<<dim3(B_ * S_ * S_ / 256), blk, 0, stream>>>(amask, mbits);

  const dim3 ggrid(D_ / 128, M_ / 128);   // (8, 32)
  gemm_awt_kernel<<<ggrid, blk, 0, stream>>>(embedH, WqH, bq, nullptr, Qh, 1);
  gemm_awt_kernel<<<ggrid, blk, 0, stream>>>(keyH,   WkH, bk, nullptr, Kh, 1);
  gemm_awt_kernel<<<ggrid, blk, 0, stream>>>(keyH,   WvH, bv, nullptr, Vt, 2);

  // attention: B * H * (S/128) = 512 blocks
  attn_flash_kernel<<<dim3(B_ * H_ * (S_ / 128)), blk, 0, stream>>>(Qh, Kh, Vt, mbits, OcH);

  // output projection
  gemm_awt_kernel<<<ggrid, blk, 0, stream>>>(OcH, WoH, bo, out, nullptr, 0);
}
